// LinearMoE_3865470566680
// MI455X (gfx1250) — matrix operand order repaired and checked by reading, not compiled, not run on hardware
//
#include <hip/hip_runtime.h>
#include <hip/hip_bf16.h>
#include <stdint.h>

// Problem sizes (match reference)
#define BB 4096
#define DD 2048
#define HH 2048
#define EE 8

// Fallback-path tiling
#define TILE_M 128
#define TILE_N 128
#define TILE_K 32
#define LDS_STRIDE 40   // bf16 elems per LDS row (80B = 64B data + 16B pad)

// Fast-path tiling (double-buffered async/TDM staging)
#define FK   64         // K-step
#define FSTR 72         // bf16 elems per LDS row (144B = 128B data + 16B pad)
#define FNK  (DD / FK)  // 32 K-iterations

typedef __bf16 v16bf  __attribute__((ext_vector_type(16)));
typedef __bf16 bf16x8 __attribute__((ext_vector_type(8)));
typedef __bf16 bf16x4 __attribute__((ext_vector_type(4)));
typedef float  v8f    __attribute__((ext_vector_type(8)));
typedef unsigned int uint32x4 __attribute__((ext_vector_type(4)));
typedef int          int32x4  __attribute__((ext_vector_type(4)));
typedef int          int32x8  __attribute__((ext_vector_type(8)));

// ---- CDNA5 async / TDM helpers ---------------------------------------------
__device__ __forceinline__ uint32_t lds_off(const void* p) {
    // generic pointer to LDS: low 32 bits are the wave-relative LDS byte offset
    return (uint32_t)(uintptr_t)p;
}
__device__ __forceinline__ void async_copy_b128(uint32_t ldsoff, const void* g) {
    uint64_t ga = (uint64_t)(uintptr_t)g;
    asm volatile("global_load_async_to_lds_b128 %0, %1, off"
                 :: "v"(ldsoff), "v"(ga) : "memory");
}
template <int N>
__device__ __forceinline__ void wait_async_le() {
    asm volatile("s_wait_asynccnt %0" :: "i"(N) : "memory");
}
__device__ __forceinline__ void tdm_load_tile_b(uint32_t ldsoff, const void* gtile) {
    // D#: 2D bf16 tensor, tile = 64 (dim0=k) x 128 (dim1=h), row stride DD elems,
    // LDS padding: 16B after every 128B -> LDS row pitch 144B (= FSTR bf16)
    uint64_t ga = (uint64_t)(uintptr_t)gtile;
    uint32x4 g0;
    g0.x = 1u;                                   // count=1 (valid descriptor)
    g0.y = ldsoff;                               // lds_addr
    g0.z = (uint32_t)ga;                         // global_addr[31:0]
    g0.w = (uint32_t)(ga >> 32) | (2u << 30);    // global_addr[56:32] | type=2
    int32x8 g1;
    g1[0] = (int)((1u << 16)                     // data_size = 2B
                | (1u << 20)                     // pad_enable
                | (4u << 22)                     // pad_interval: 32 DWORDs (128B)
                | (3u << 25));                   // pad_amount: 4 DWORDs (16B)
    g1[1] = (int)((DD & 0xFFFFu) << 16);         // tensor_dim0[15:0] @ bits 63:48
    g1[2] = (int)(((DD >> 16) & 0xFFFFu) | ((HH & 0xFFFFu) << 16)); // dim0 hi | dim1 lo
    g1[3] = (int)(((HH >> 16) & 0xFFFFu) | ((unsigned)FK << 16));   // dim1 hi | tile_dim0
    g1[4] = (int)(128u);                         // tile_dim1=128, tile_dim2=0
    g1[5] = (int)(DD);                           // tensor_dim0_stride (elements)
    g1[6] = 0;
    g1[7] = 0;
    int32x4 z4 = (int32x4)(0);
#if defined(__clang_major__) && (__clang_major__ >= 23)
    int32x8 z8 = (int32x8)(0);
    __builtin_amdgcn_tensor_load_to_lds(g0, g1, z4, z4, z8, 0);
#else
    __builtin_amdgcn_tensor_load_to_lds(g0, g1, z4, z4, 0);
#endif
}
// One tile stage: per-thread 64B of A (4x async b128, gathered rows) + wave0 TDM for B
__device__ __forceinline__ void stage_tile(const __bf16* a_src, uint32_t a_lds,
                                           const __bf16* b_src, uint32_t b_lds, int wave) {
    async_copy_b128(a_lds,      a_src);
    async_copy_b128(a_lds + 16, a_src + 8);
    async_copy_b128(a_lds + 32, a_src + 16);
    async_copy_b128(a_lds + 48, a_src + 24);
    if (wave == 0) tdm_load_tile_b(b_lds, b_src);
}

// ---------------------------------------------------------------------------
// Kernel 0: zero output + per-expert counters
// ---------------------------------------------------------------------------
__global__ __launch_bounds__(256) void moe_zero(float4* __restrict__ out4, int n4,
                                                int* __restrict__ cnt) {
    int i = blockIdx.x * blockDim.x + threadIdx.x;
    if (i < n4) out4[i] = make_float4(0.f, 0.f, 0.f, 0.f);
    if (i < EE) cnt[i] = 0;
}

// ---------------------------------------------------------------------------
// Kernel 1: gating = softmax(x @ Wg + bg), top-2, scatter into expert lists.
// ---------------------------------------------------------------------------
__global__ __launch_bounds__(256) void moe_gating(const float* __restrict__ x,
                                                  const float* __restrict__ Wg,
                                                  const float* __restrict__ bg,
                                                  int*   __restrict__ cnt,
                                                  int*   __restrict__ tok,
                                                  float* __restrict__ wgt) {
    const int lane = threadIdx.x & 31;
    const int wave = threadIdx.x >> 5;
    const int t = blockIdx.x * 8 + wave;
    if (t >= BB) return;

    float acc[EE];
#pragma unroll
    for (int e = 0; e < EE; ++e) acc[e] = 0.f;

    const float* xr = x + (size_t)t * DD;
    for (int d = lane; d < DD; d += 32) {
        float xv = xr[d];
        const float4* wr = (const float4*)(Wg + (size_t)d * EE);
        float4 w0 = wr[0];
        float4 w1 = wr[1];
        acc[0] += xv * w0.x; acc[1] += xv * w0.y;
        acc[2] += xv * w0.z; acc[3] += xv * w0.w;
        acc[4] += xv * w1.x; acc[5] += xv * w1.y;
        acc[6] += xv * w1.z; acc[7] += xv * w1.w;
    }
#pragma unroll
    for (int off = 16; off > 0; off >>= 1) {
#pragma unroll
        for (int e = 0; e < EE; ++e)
            acc[e] += __shfl_xor(acc[e], off, 32);
    }

    if (lane == 0) {
        float l[EE];
#pragma unroll
        for (int e = 0; e < EE; ++e) l[e] = acc[e] + bg[e];
        float m = l[0];
#pragma unroll
        for (int e = 1; e < EE; ++e) m = fmaxf(m, l[e]);
        float s = 0.f, g[EE];
#pragma unroll
        for (int e = 0; e < EE; ++e) { g[e] = __expf(l[e] - m); s += g[e]; }
        float inv = 1.f / s;
#pragma unroll
        for (int e = 0; e < EE; ++e) g[e] *= inv;

        int i1 = 0; float g1 = g[0];
#pragma unroll
        for (int e = 1; e < EE; ++e) if (g[e] > g1) { g1 = g[e]; i1 = e; }
        int i2 = -1; float g2 = -1.f;
#pragma unroll
        for (int e = 0; e < EE; ++e) if (e != i1 && g[e] > g2) { g2 = g[e]; i2 = e; }

        int p1 = atomicAdd(&cnt[i1], 1);
        tok[i1 * BB + p1] = t;
        wgt[i1 * BB + p1] = g1;
        int p2 = atomicAdd(&cnt[i2], 1);
        tok[i2 * BB + p2] = t;
        wgt[i2 * BB + p2] = g2;
    }
}

// ---------------------------------------------------------------------------
// Conversion kernels (fast path): x -> bf16, We -> transposed bf16 WeT[e][h][d]
// ---------------------------------------------------------------------------
__global__ __launch_bounds__(256) void cvt_x_bf16(const float* __restrict__ x,
                                                  __bf16* __restrict__ xb, int n8) {
    int i = blockIdx.x * blockDim.x + threadIdx.x;
    if (i >= n8) return;
    const float4* s = (const float4*)(x + (size_t)i * 8);
    float4 a = s[0], b = s[1];
    bf16x8 o;
    o[0] = (__bf16)a.x; o[1] = (__bf16)a.y; o[2] = (__bf16)a.z; o[3] = (__bf16)a.w;
    o[4] = (__bf16)b.x; o[5] = (__bf16)b.y; o[6] = (__bf16)b.z; o[7] = (__bf16)b.w;
    *(bf16x8*)(xb + (size_t)i * 8) = o;
}

// grid (HH/64, DD/64, EE), 256 threads: 64x64 tile transpose via LDS
__global__ __launch_bounds__(256) void cvt_we_transpose(const float* __restrict__ We,
                                                        __bf16* __restrict__ WeT) {
    const int e  = blockIdx.z;
    const int h0 = blockIdx.x * 64;
    const int d0 = blockIdx.y * 64;
    __shared__ float t[64][65];

    const float* src = We + (size_t)e * DD * HH;
    const int r = threadIdx.x >> 4;
    const int c = (threadIdx.x & 15) * 4;
#pragma unroll
    for (int s4 = 0; s4 < 4; ++s4) {
        int d = r + s4 * 16;
        float4 v = *(const float4*)(src + (size_t)(d0 + d) * HH + h0 + c);
        t[d][c + 0] = v.x; t[d][c + 1] = v.y; t[d][c + 2] = v.z; t[d][c + 3] = v.w;
    }
    __syncthreads();
    __bf16* dst = WeT + (size_t)e * DD * HH;
#pragma unroll
    for (int s4 = 0; s4 < 4; ++s4) {
        int h = r + s4 * 16;
        bf16x4 o;
        o[0] = (__bf16)t[c + 0][h]; o[1] = (__bf16)t[c + 1][h];
        o[2] = (__bf16)t[c + 2][h]; o[3] = (__bf16)t[c + 3][h];
        *(bf16x4*)(dst + (size_t)(h0 + h) * DD + d0 + c) = o;
    }
}

// ---------------------------------------------------------------------------
// Kernel 2 (fast path): routed grouped GEMM, bf16 operands, double-buffered
// async/TDM staging pipelined against WMMA compute.
// ---------------------------------------------------------------------------
__global__ __launch_bounds__(256) void moe_expert_gemm_bf16(
        const __bf16* __restrict__ xb, const __bf16* __restrict__ WeT,
        const float* __restrict__ be, const int* __restrict__ cnt,
        const int* __restrict__ tok, const float* __restrict__ wgt,
        float* __restrict__ out) {
    const int e     = blockIdx.z;
    const int count = cnt[e];
    const int m0    = blockIdx.y * TILE_M;
    if (m0 >= count) return;
    const int n0    = blockIdx.x * TILE_N;

    __shared__ __bf16 As[2][TILE_M * FSTR];
    __shared__ __bf16 Bs[2][TILE_N * FSTR];
    __shared__ int    sTok[TILE_M];
    __shared__ float  sW[TILE_M];

    const int tid  = threadIdx.x;
    const int lane = tid & 31;
    const int wave = tid >> 5;

    if (tid < TILE_M) {
        int m = m0 + tid;
        if (m < count) { sTok[tid] = tok[e * BB + m]; sW[tid] = wgt[e * BB + m]; }
        else           { sTok[tid] = 0;               sW[tid] = 0.f; }
    }
    __syncthreads();

    // A async mapping: 2 threads per row, 32 bf16 (64B) each = 4x b128 async
    const int a_row  = tid >> 1;
    const int a_half = (tid & 1) * 32;
    const __bf16* a_src = xb + (size_t)sTok[a_row] * DD + a_half;
    const uint32_t a_lds[2] = { lds_off(&As[0][a_row * FSTR + a_half]),
                                lds_off(&As[1][a_row * FSTR + a_half]) };
    const uint32_t b_lds[2] = { lds_off(&Bs[0][0]), lds_off(&Bs[1][0]) };
    const __bf16* __restrict__ Wexp = WeT + (size_t)e * DD * HH + (size_t)n0 * DD;

    v8f acc[8];
#pragma unroll
    for (int mb = 0; mb < 8; ++mb) acc[mb] = (v8f)(0.f);

    const int kh = lane >> 4;
    const int ml = lane & 15;

    // prologue: stage tile 0 into buffer 0
    stage_tile(a_src, a_lds[0], Wexp, b_lds[0], wave);

    for (int it = 0; it < FNK; ++it) {
        const int p = it & 1;
        if (it + 1 < FNK) {
            // issue tile it+1 into the other buffer (overlaps with compute below)
            stage_tile(a_src + (it + 1) * FK, a_lds[p ^ 1],
                       Wexp + (it + 1) * FK,  b_lds[p ^ 1], wave);
            wait_async_le<4>();                      // tile it's 4 copies done (in-order)
            __builtin_amdgcn_s_wait_tensorcnt(1);    // tile it's TDM done (in-order)
        } else {
            wait_async_le<0>();
            __builtin_amdgcn_s_wait_tensorcnt(0);
        }
        __syncthreads();   // all waves see whole tile `it`

        // compute on buffer p: 2 K-sub-steps x 8 M-blocks = 16 WMMAs
#pragma unroll
        for (int ks = 0; ks < 2; ++ks) {
            v16bf bfrag;
            {
                const __bf16* bp = &Bs[p][(wave * 16 + ml) * FSTR + ks * 32 + kh * 8];
                bf16x8 lo = *(const bf16x8*)(bp);
                bf16x8 hi = *(const bf16x8*)(bp + 16);
#pragma unroll
                for (int i = 0; i < 8; ++i) { bfrag[i] = lo[i]; bfrag[8 + i] = hi[i]; }
            }
#pragma unroll
            for (int mb = 0; mb < 8; ++mb) {
                const __bf16* ap = &As[p][(mb * 16 + ml) * FSTR + ks * 32 + kh * 8];
                bf16x8 lo = *(const bf16x8*)(ap);        // K = ks*32 + 8*kh + 0..7
                bf16x8 hi = *(const bf16x8*)(ap + 16);   // K = ks*32 + 16 + 8*kh + 0..7
                v16bf afrag;
#pragma unroll
                for (int i = 0; i < 8; ++i) { afrag[i] = lo[i]; afrag[8 + i] = hi[i]; }
                acc[mb] = __builtin_amdgcn_wmma_f32_16x16x32_bf16(
                              false, afrag, false, bfrag,
                              (short)0, acc[mb], false, false);
            }
        }
        __syncthreads();   // buffer p free before tile it+2 streams into it
    }

    const int hcol = n0 + wave * 16 + ml;
    const float bcol = be[e * HH + hcol];
#pragma unroll
    for (int mb = 0; mb < 8; ++mb) {
#pragma unroll
        for (int r = 0; r < 8; ++r) {
            int m = mb * 16 + r + 8 * kh;
            float w = sW[m];
            int   t = sTok[m];
            float v = (acc[mb][r] + bcol) * w;
            atomicAdd(&out[(size_t)t * HH + hcol], v);
        }
    }
}

// ---------------------------------------------------------------------------
// Kernel 2 (fallback): f32 inputs converted on the fly
// ---------------------------------------------------------------------------
__global__ __launch_bounds__(256) void moe_expert_gemm_f32(
        const float* __restrict__ x, const float* __restrict__ We,
        const float* __restrict__ be, const int* __restrict__ cnt,
        const int* __restrict__ tok, const float* __restrict__ wgt,
        float* __restrict__ out) {
    const int e     = blockIdx.z;
    const int count = cnt[e];
    const int m0    = blockIdx.y * TILE_M;
    if (m0 >= count) return;
    const int n0    = blockIdx.x * TILE_N;

    __shared__ __bf16 As[TILE_M * LDS_STRIDE];
    __shared__ __bf16 Bs[TILE_N * LDS_STRIDE];
    __shared__ int    sTok[TILE_M];
    __shared__ float  sW[TILE_M];

    const int tid  = threadIdx.x;
    const int lane = tid & 31;
    const int wave = tid >> 5;

    if (tid < TILE_M) {
        int m = m0 + tid;
        if (m < count) { sTok[tid] = tok[e * BB + m]; sW[tid] = wgt[e * BB + m]; }
        else           { sTok[tid] = 0;               sW[tid] = 0.f; }
    }
    __syncthreads();

    const int a_row = tid >> 1;
    const int a_col = (tid & 1) * 16;
    const int a_tok = sTok[a_row];
    const int b_k = tid >> 3;
    const int b_c = (tid & 7) * 16;

    const float* __restrict__ Wexp = We + (size_t)e * DD * HH;

    v8f acc[8];
#pragma unroll
    for (int mb = 0; mb < 8; ++mb) acc[mb] = (v8f)(0.f);

    const int kh = lane >> 4;
    const int ml = lane & 15;

    for (int k0 = 0; k0 < DD; k0 += TILE_K) {
        {
            const float4* src = (const float4*)(x + (size_t)a_tok * DD + k0 + a_col);
#pragma unroll
            for (int j = 0; j < 4; ++j) {
                float4 v = src[j];
                __bf16* dst = &As[a_row * LDS_STRIDE + a_col + j * 4];
                dst[0] = (__bf16)v.x; dst[1] = (__bf16)v.y;
                dst[2] = (__bf16)v.z; dst[3] = (__bf16)v.w;
            }
        }
        {
            const float4* bsrc = (const float4*)(Wexp + (size_t)(k0 + b_k) * HH + n0 + b_c);
#pragma unroll
            for (int j = 0; j < 4; ++j) {
                float4 v = bsrc[j];
                int c = b_c + j * 4;
                Bs[(c + 0) * LDS_STRIDE + b_k] = (__bf16)v.x;
                Bs[(c + 1) * LDS_STRIDE + b_k] = (__bf16)v.y;
                Bs[(c + 2) * LDS_STRIDE + b_k] = (__bf16)v.z;
                Bs[(c + 3) * LDS_STRIDE + b_k] = (__bf16)v.w;
            }
            if (k0 + TILE_K < DD)
                __builtin_prefetch(Wexp + (size_t)(k0 + TILE_K + b_k) * HH + n0 + b_c, 0, 0);
        }
        __syncthreads();

        v16bf bfrag;
        {
            const __bf16* p = &Bs[(wave * 16 + ml) * LDS_STRIDE + kh * 8];
            bf16x8 lo = *(const bf16x8*)(p);
            bf16x8 hi = *(const bf16x8*)(p + 16);
#pragma unroll
            for (int i = 0; i < 8; ++i) { bfrag[i] = lo[i]; bfrag[8 + i] = hi[i]; }
        }
#pragma unroll
        for (int mb = 0; mb < 8; ++mb) {
            const __bf16* p = &As[(mb * 16 + ml) * LDS_STRIDE + kh * 8];
            bf16x8 lo = *(const bf16x8*)(p);
            bf16x8 hi = *(const bf16x8*)(p + 16);
            v16bf afrag;
#pragma unroll
            for (int i = 0; i < 8; ++i) { afrag[i] = lo[i]; afrag[8 + i] = hi[i]; }
            acc[mb] = __builtin_amdgcn_wmma_f32_16x16x32_bf16(
                          false, afrag, false, bfrag,
                          (short)0, acc[mb], false, false);
        }
        __syncthreads();
    }

    const int hcol = n0 + wave * 16 + ml;
    const float bcol = be[e * HH + hcol];
#pragma unroll
    for (int mb = 0; mb < 8; ++mb) {
#pragma unroll
        for (int r = 0; r < 8; ++r) {
            int m = mb * 16 + r + 8 * kh;
            float w = sW[m];
            int   t = sTok[m];
            float v = (acc[mb][r] + bcol) * w;
            atomicAdd(&out[(size_t)t * HH + hcol], v);
        }
    }
}

// ---------------------------------------------------------------------------
extern "C" void kernel_launch(void* const* d_in, const int* in_sizes, int n_in,
                              void* d_out, int out_size, void* d_ws, size_t ws_size,
                              hipStream_t stream) {
    const float* x  = (const float*)d_in[0];
    const float* Wg = (const float*)d_in[1];
    const float* bg = (const float*)d_in[2];
    const float* We = (const float*)d_in[3];
    const float* be = (const float*)d_in[4];
    float* out = (float*)d_out;

    char* ws = (char*)d_ws;
    int*   cnt = (int*)ws;
    int*   tok = (int*)(ws + 1024);
    float* wgt = (float*)(ws + 1024 + (size_t)EE * BB * sizeof(int));

    const size_t route_bytes = 1024 + (size_t)EE * BB * 4 * 2;
    const size_t xb_bytes    = (size_t)BB * DD * 2;
    const size_t weT_bytes   = (size_t)EE * DD * HH * 2;
    const size_t need_fast   = route_bytes + xb_bytes + weT_bytes;

    const int n4 = (BB * HH) / 4;
    moe_zero<<<(n4 + 255) / 256, 256, 0, stream>>>((float4*)out, n4, cnt);
    moe_gating<<<BB / 8, 256, 0, stream>>>(x, Wg, bg, cnt, tok, wgt);

    if (ws_size >= need_fast) {
        __bf16* xb  = (__bf16*)(ws + route_bytes);
        __bf16* weT = (__bf16*)(ws + route_bytes + xb_bytes);
        cvt_x_bf16<<<(BB * DD / 8 + 255) / 256, 256, 0, stream>>>(x, xb, BB * DD / 8);
        dim3 tg(HH / 64, DD / 64, EE);
        cvt_we_transpose<<<tg, 256, 0, stream>>>(We, weT);
        dim3 grid(HH / TILE_N, BB / TILE_M, EE);
        moe_expert_gemm_bf16<<<grid, 256, 0, stream>>>(xb, weT, be, cnt, tok, wgt, out);
    } else {
        dim3 grid(HH / TILE_N, BB / TILE_M, EE);
        moe_expert_gemm_f32<<<grid, 256, 0, stream>>>(x, We, be, cnt, tok, wgt, out);
    }
}
